// OTAlignModule_25941602468364
// MI455X (gfx1250) — matrix operand order repaired and checked by reading, not compiled, not run on hardware
//
#include <hip/hip_runtime.h>

// ---------------------------------------------------------------------------
// OT-align (Sinkhorn) for MI455X / gfx1250, wave32 + WMMA bf16.
//   B=4, C=256, H=W=48 -> M=2304, N=4096, REG=10, ALPHA=1e-3, 200 iters.
// K stored bf16 (75.5MB -> L2-resident on 192MB L2); both GEMMs use
// v_wmma_f32_16x16x32_bf16 with pre-swizzled fragment-ready operands.
// ---------------------------------------------------------------------------

typedef __attribute__((ext_vector_type(16))) __bf16 v16bf;
typedef __attribute__((ext_vector_type(8)))  float  v8f;

#define BN 4
#define CC 256
#define MM 2304      // 48*48
#define NN 4096
#define TM 144       // MM/16
#define TN 256       // NN/16
#define TC 16        // CC/16

__device__ __forceinline__ float bf2f(unsigned short h) {
  return __uint_as_float(((unsigned)h) << 16);
}
__device__ __forceinline__ unsigned short f2bf(float f) {
  unsigned u = __float_as_uint(f);
  u += 0x7fffu + ((u >> 16) & 1u);          // round-to-nearest-even
  return (unsigned short)(u >> 16);
}
// WMMA A-matrix (16x32 bf16) lane/half -> K index
__device__ __forceinline__ int amap(int lane, int j) {
  return ((lane >> 4) << 4) + j;
}
// WMMA B-matrix (32x16 bf16) lane/half -> K index
__device__ __forceinline__ int bmapk(int lane, int j) {
  return ((lane >> 4) << 4) + j;
}
__device__ __forceinline__ void store16h(unsigned short* p, const unsigned short* hv) {
  uint4 q0, q1;
  q0.x = (unsigned)hv[0]  | ((unsigned)hv[1]  << 16);
  q0.y = (unsigned)hv[2]  | ((unsigned)hv[3]  << 16);
  q0.z = (unsigned)hv[4]  | ((unsigned)hv[5]  << 16);
  q0.w = (unsigned)hv[6]  | ((unsigned)hv[7]  << 16);
  q1.x = (unsigned)hv[8]  | ((unsigned)hv[9]  << 16);
  q1.y = (unsigned)hv[10] | ((unsigned)hv[11] << 16);
  q1.z = (unsigned)hv[12] | ((unsigned)hv[13] << 16);
  q1.w = (unsigned)hv[14] | ((unsigned)hv[15] << 16);
  ((uint4*)p)[0] = q0;
  ((uint4*)p)[1] = q1;
}

// ---------- pack kernels: build fragment-ready bf16 operand buffers ----------

// A-fragments of X (rows n, k = c): frag[t_n][kc][lane][16]
__global__ __launch_bounds__(256) void pack_XhA(const float* __restrict__ sc,
                                                unsigned short* __restrict__ XhA) {
  int t = blockIdx.x * 256 + threadIdx.x;          // 65536 lane-frags
  if (t >= TN * 8 * 32) return;
  int L = t & 31, r = t >> 5;
  int kc = r & 7, t_n = r >> 3;
  int n = t_n * 16 + (L & 15);
  unsigned short hv[16];
#pragma unroll
  for (int j = 0; j < 16; ++j) {
    int c = kc * 32 + amap(L, j);
    hv[j] = f2bf(sc[n * 256 + c]);
  }
  store16h(XhA + (size_t)t * 16, hv);
}

// A-fragments of X^T (rows c, k = n): frag[t_c][kc_n][lane][16]
__global__ __launch_bounds__(256) void pack_XTA(const float* __restrict__ sc,
                                                unsigned short* __restrict__ XTA) {
  int t = blockIdx.x * 256 + threadIdx.x;          // 65536 lane-frags
  if (t >= TC * 128 * 32) return;
  int L = t & 31, r = t >> 5;
  int kc = r & 127, t_c = r >> 7;
  int c = t_c * 16 + (L & 15);
  unsigned short hv[16];
#pragma unroll
  for (int j = 0; j < 16; ++j) {
    int n = kc * 32 + amap(L, j);
    hv[j] = f2bf(sc[n * 256 + c]);
  }
  store16h(XTA + (size_t)t * 16, hv);
}

// B-fragments of Y^T (k = c, cols m): frag[b][t_m][kc][lane][16]
__global__ __launch_bounds__(256) void pack_YhB(const float* __restrict__ h,
                                                unsigned short* __restrict__ YhB) {
  int t = blockIdx.x * 256 + threadIdx.x;          // 147456 lane-frags
  if (t >= BN * TM * 8 * 32) return;
  int L = t & 31, r = t >> 5;
  int kc = r & 7; r >>= 3;
  int t_m = r % TM, b = r / TM;
  int m = t_m * 16 + (L & 15);
  unsigned short hv[16];
#pragma unroll
  for (int j = 0; j < 16; ++j) {
    int c = kc * 32 + bmapk(L, j);
    hv[j] = f2bf(h[((size_t)(b * 256 + c)) * MM + m]);
  }
  store16h(YhB + (size_t)t * 16, hv);
}

// ---------- norms ----------

__global__ __launch_bounds__(256) void calc_x2(const float* __restrict__ sc,
                                               float* __restrict__ x2) {
  int lane = threadIdx.x & 31, wid = threadIdx.x >> 5;
  int n = blockIdx.x * 8 + wid;                    // grid 512
  const float* row = sc + (size_t)n * 256;
  float s = 0.f;
#pragma unroll
  for (int i = 0; i < 8; ++i) { float v = row[lane + i * 32]; s += v * v; }
#pragma unroll
  for (int off = 16; off; off >>= 1) s += __shfl_xor(s, off, 32);
  if (lane == 0) x2[n] = s;
}

__global__ __launch_bounds__(256) void calc_y2(const float* __restrict__ h,
                                               float* __restrict__ y2) {
  int t = blockIdx.x * 256 + threadIdx.x;          // 9216
  if (t >= BN * MM) return;
  int b = t / MM, m = t % MM;
  float s = 0.f;
  for (int c = 0; c < 256; ++c) {
    float v = h[((size_t)(b * 256 + c)) * MM + m];
    s += v * v;
  }
  y2[t] = s;
}

// ---------- GEMM 1: cost + K = exp(-cost/REG) (bf16 out) ----------
// block = 128 (4 waves); wave -> one 16x16 tile; grid (TN, TM/4, BN)
__global__ __launch_bounds__(128) void gemm_cost(const unsigned short* __restrict__ XhA,
                                                 const unsigned short* __restrict__ YhB,
                                                 const float* __restrict__ x2,
                                                 const float* __restrict__ y2,
                                                 unsigned short* __restrict__ Kbf) {
  int lane = threadIdx.x & 31, wid = threadIdx.x >> 5;
  int t_n = blockIdx.x, t_m = blockIdx.y * 4 + wid, b = blockIdx.z;
  const v16bf* Av = reinterpret_cast<const v16bf*>(XhA) + (size_t)(t_n * 8) * 32 + lane;
  const v16bf* Bv = reinterpret_cast<const v16bf*>(YhB) + (size_t)((b * TM + t_m) * 8) * 32 + lane;
  v8f acc = {0.f, 0.f, 0.f, 0.f, 0.f, 0.f, 0.f, 0.f};
#pragma unroll
  for (int kc = 0; kc < 8; ++kc) {
    v16bf a  = Av[kc * 32];
    v16bf bb = Bv[kc * 32];
    acc = __builtin_amdgcn_wmma_f32_16x16x32_bf16(false, a, false, bb,
                                                  (short)0, acc, false, false);
  }
  int col = lane & 15, rowb = (lane >> 4) * 8;
  int m = t_m * 16 + col;
  float y2v = y2[b * MM + m];
  size_t baseK = ((size_t)b * NN + t_n * 16) * MM + m;
#pragma unroll
  for (int r = 0; r < 8; ++r) {
    int n = t_n * 16 + rowb + r;
    float cost = x2[n] + y2v - 2.0f * acc[r];
    cost = fmaxf(cost, 0.0f);
    float kv = expf(cost * -0.1f);                 // exp(-cost/REG), REG=10
    Kbf[baseK + (size_t)(rowb + r) * MM] = f2bf(kv);
  }
}

// ---------- Sinkhorn ----------

__global__ __launch_bounds__(256) void init_u(float* __restrict__ u) {
  int t = blockIdx.x * 256 + threadIdx.x;          // 16384
  if (t < BN * NN) u[t] = 1.0f / (float)NN;
}

// partial column sums: part[chunk*4+b][m] = sum_{n in chunk} K[b,n,m]*u[b,n]
__global__ __launch_bounds__(256) void sink_colsum(const unsigned short* __restrict__ Kbf,
                                                   const float* __restrict__ u,
                                                   float* __restrict__ part) {
  int m = blockIdx.x * 256 + threadIdx.x;          // grid (9, 8, 4)
  int chunk = blockIdx.y, b = blockIdx.z;
  int n0 = chunk * 512;
  const unsigned short* Kp = Kbf + ((size_t)b * NN + n0) * MM + m;
  const float* up = u + b * NN + n0;
  float s = 0.f;
  for (int i = 0; i < 512; ++i)
    s += bf2f(Kp[(size_t)i * MM]) * up[i];         // u uniform -> scalar loads
  part[(size_t)(chunk * BN + b) * MM + m] = s;
}

__global__ __launch_bounds__(256) void sink_vcalc(const float* __restrict__ part,
                                                  float* __restrict__ v) {
  int t = blockIdx.x * 256 + threadIdx.x;          // 9216
  if (t >= BN * MM) return;
  int b = t / MM, m = t % MM;
  float s = 0.f;
#pragma unroll
  for (int ch = 0; ch < 8; ++ch) s += part[(size_t)(ch * BN + b) * MM + m];
  v[t] = (1.0f / (float)MM) / s;
}

// row sums: u[b,n] = (1/N) / sum_m K[b,n,m]*v[b,m];  one wave per row
__global__ __launch_bounds__(256) void sink_rowsum(const unsigned short* __restrict__ Kbf,
                                                   const float* __restrict__ v,
                                                   float* __restrict__ u) {
  __shared__ float vl[MM];
  int lane = threadIdx.x & 31, wid = threadIdx.x >> 5;
  int b = blockIdx.x >> 9;                         // 4096 rows/batch, 8 rows/block
  for (int i = threadIdx.x; i < MM; i += 256) vl[i] = v[b * MM + i];
  __syncthreads();
  int n = (blockIdx.x * 8 + wid) & (NN - 1);
  const unsigned short* row = Kbf + ((size_t)b * NN + n) * MM;
  float s = 0.f;
#pragma unroll
  for (int i = 0; i < 9; ++i) {
    int base = i * 256 + lane * 8;
    uint4 q = *reinterpret_cast<const uint4*>(row + base);
    s += bf2f(q.x & 0xffff) * vl[base + 0] + bf2f(q.x >> 16) * vl[base + 1];
    s += bf2f(q.y & 0xffff) * vl[base + 2] + bf2f(q.y >> 16) * vl[base + 3];
    s += bf2f(q.z & 0xffff) * vl[base + 4] + bf2f(q.z >> 16) * vl[base + 5];
    s += bf2f(q.w & 0xffff) * vl[base + 6] + bf2f(q.w >> 16) * vl[base + 7];
  }
#pragma unroll
  for (int off = 16; off; off >>= 1) s += __shfl_xor(s, off, 32);
  if (lane == 0) u[b * NN + n] = (1.0f / (float)NN) / s;
}

// ---------- pack B-fragments of (u o K): frag[b][kc_n][t_m][lane][16] ----------
__global__ __launch_bounds__(256) void pack_UKt(const unsigned short* __restrict__ Kbf,
                                                const float* __restrict__ u,
                                                unsigned short* __restrict__ UKt) {
  int t = blockIdx.x * 256 + threadIdx.x;          // 2,359,296 lane-frags
  if (t >= BN * 128 * TM * 32) return;
  int L = t & 31, r = t >> 5;
  int t_m = r % TM; r /= TM;
  int kc = r & 127, b = r >> 7;
  int m = t_m * 16 + (L & 15);
  unsigned short hv[16];
#pragma unroll
  for (int j = 0; j < 16; ++j) {
    int n = kc * 32 + bmapk(L, j);
    float val = u[b * NN + n] * bf2f(Kbf[((size_t)b * NN + n) * MM + m]);
    hv[j] = f2bf(val);
  }
  store16h(UKt + (size_t)t * 16, hv);
}

// ---------- GEMM 2: ot_feat[b,c,m] = v[m] * sum_n X[n,c]*u[n]*K[n,m] ----------
__global__ __launch_bounds__(128) void gemm_out(const unsigned short* __restrict__ XTA,
                                                const unsigned short* __restrict__ UKt,
                                                const float* __restrict__ v,
                                                float* __restrict__ ot) {
  int lane = threadIdx.x & 31, wid = threadIdx.x >> 5;
  int t_c = blockIdx.x, t_m = blockIdx.y * 4 + wid, b = blockIdx.z;
  const v16bf* Av = reinterpret_cast<const v16bf*>(XTA) + (size_t)(t_c * 128) * 32 + lane;
  const v16bf* Bv = reinterpret_cast<const v16bf*>(UKt) +
                    ((size_t)(b * 128) * TM + t_m) * 32 + lane;
  v8f acc = {0.f, 0.f, 0.f, 0.f, 0.f, 0.f, 0.f, 0.f};
#pragma unroll 8
  for (int kc = 0; kc < 128; ++kc) {
    v16bf a  = Av[kc * 32];
    v16bf bb = Bv[(size_t)kc * (TM * 32)];
    acc = __builtin_amdgcn_wmma_f32_16x16x32_bf16(false, a, false, bb,
                                                  (short)0, acc, false, false);
  }
  int col = lane & 15, rowb = (lane >> 4) * 8;
  int m = t_m * 16 + col;
  float vm = v[b * MM + m];
#pragma unroll
  for (int r = 0; r < 8; ++r) {
    int c = t_c * 16 + rowb + r;
    ot[((size_t)(b * 256 + c)) * MM + m] = acc[r] * vm;
  }
}

// ---------- outputs ----------

__global__ void zero_loss(float* __restrict__ out) { out[2359296] = 0.0f; }

__global__ __launch_bounds__(256) void finalize(const float* __restrict__ h,
                                                const float* __restrict__ ot,
                                                float* __restrict__ out) {
  int i = blockIdx.x * 256 + threadIdx.x;          // 2,359,296 exact
  float hv = h[i];
  float ov = ot[i];
  out[i] = hv + 0.001f * ov;                       // h_spot_aligned
  out[2359297 + i] = ov;                           // ot_feat
  float d = hv - ov;
  float s = d * d;
#pragma unroll
  for (int off = 16; off; off >>= 1) s += __shfl_xor(s, off, 32);
  __shared__ float red[8];
  int lane = threadIdx.x & 31, wid = threadIdx.x >> 5;
  if (lane == 0) red[wid] = s;
  __syncthreads();
  if (threadIdx.x == 0) {
    float t = 0.f;
#pragma unroll
    for (int k = 0; k < 8; ++k) t += red[k];
    atomicAdd(out + 2359296, t * (1.0f / 2359296.0f));
  }
}

// ---------------------------------------------------------------------------

extern "C" void kernel_launch(void* const* d_in, const int* in_sizes, int n_in,
                              void* d_out, int out_size, void* d_ws, size_t ws_size,
                              hipStream_t stream) {
  const float* sc = (const float*)d_in[0];         // (4096, 256)
  const float* h  = (const float*)d_in[1];         // (4, 256, 48, 48)
  float* out = (float*)d_out;                      // 2*2359296 + 1 floats

  char* ws = (char*)d_ws;
  unsigned short* Kbf = (unsigned short*)(ws + 0);             // 75,497,472 B
  unsigned short* XhA = (unsigned short*)(ws + 75497472u);     //  2,097,152 B
  unsigned short* XTA = (unsigned short*)(ws + 77594624u);     //  2,097,152 B
  float* x2   = (float*)(ws + 79691776u);                      //     16,384 B
  float* y2   = (float*)(ws + 79708160u);                      //     36,864 B
  float* u    = (float*)(ws + 79745024u);                      //     65,536 B
  float* v    = (float*)(ws + 79810560u);                      //     36,864 B
  float* part = (float*)(ws + 79847424u);                      //    294,912 B
  float* ot   = (float*)(ws + 80142336u);                      //  9,437,184 B
  unsigned short* YhB = (unsigned short*)(ws + 89579520u);     //  4,718,592 B (dead after gemm_cost)
  unsigned short* UKt = (unsigned short*)(ws + 89579520u);     // 75,497,472 B (reuses YhB region)
  // total workspace need: 165,076,992 bytes

  // --- operand packing + norms ---
  pack_XhA<<<256, 256, 0, stream>>>(sc, XhA);
  pack_XTA<<<256, 256, 0, stream>>>(sc, XTA);
  pack_YhB<<<576, 256, 0, stream>>>(h, YhB);
  calc_x2 <<<512, 256, 0, stream>>>(sc, x2);
  calc_y2 <<< 36, 256, 0, stream>>>(h, y2);

  // --- K = exp(-cost/REG) via WMMA bf16 ---
  gemm_cost<<<dim3(TN, TM / 4, BN), 128, 0, stream>>>(XhA, YhB, x2, y2, Kbf);

  // --- Sinkhorn: 200 iterations, K resident in L2 (75.5MB of 192MB) ---
  init_u<<<64, 256, 0, stream>>>(u);
  for (int it = 0; it < 200; ++it) {
    sink_colsum<<<dim3(9, 8, BN), 256, 0, stream>>>(Kbf, u, part);
    sink_vcalc <<<36, 256, 0, stream>>>(part, v);
    sink_rowsum<<<2048, 256, 0, stream>>>(Kbf, v, u);
  }

  // --- ot_feat^T = X^T (u o K), scaled by v, via WMMA bf16 ---
  pack_UKt<<<9216, 256, 0, stream>>>(Kbf, u, UKt);
  gemm_out<<<dim3(TC, TM / 4, BN), 128, 0, stream>>>(XTA, UKt, v, ot);

  // --- outputs + MSE loss ---
  zero_loss<<<1, 1, 0, stream>>>(out);
  finalize<<<9216, 256, 0, stream>>>(h, ot, out);
}
